// GraphNetworkClassifierMag_9174050144532
// MI455X (gfx1250) — matrix operand order repaired and checked by reading, not compiled, not run on hardware
//
#include <hip/hip_runtime.h>

// ---------------------------------------------------------------------------
// CDNA5 (gfx1250) WMMA helpers: bf16 inputs, f32 accumulate, wave32.
// Layouts per CDNA5 ISA 7.12.2:
//   A (16x32 bf16): lane m=l&15, half=l>>4; elem e -> k = ((e>>3)<<4) + half*8 + (e&7)
//   B (32x16 bf16): lane n=l&15, half=l>>4; elem e -> k = ((e>>3)<<4) + half*8 + (e&7)
//   C/D (16x16 f32): elem r -> row = half*8 + r, col = l&15
// B matrices are pre-swizzled into fragment order (bf16) so each lane's 16
// elements are one contiguous 32-byte load.
// ---------------------------------------------------------------------------
typedef __attribute__((ext_vector_type(16))) __bf16 bf16x16;
typedef __attribute__((ext_vector_type(8)))  float  floatx8;

#define SQ3f      1.7320508075688772f
#define INV_SQ3f  0.5773502691896258f
#define INV_SQ2f  0.7071067811865476f

__device__ __forceinline__ floatx8 zero8() {
  floatx8 z = {0.f, 0.f, 0.f, 0.f, 0.f, 0.f, 0.f, 0.f};
  return z;
}

__device__ __forceinline__ int a_kidx(int e, int half) {
  return ((e >> 3) << 4) + half * 8 + (e & 7);
}

__device__ __forceinline__ floatx8 wmma_bf16(bf16x16 a, bf16x16 b, floatx8 c) {
  return __builtin_amdgcn_wmma_f32_16x16x32_bf16(false, a, false, b, (short)0, c,
                                                 false, false);
}

// B fragment from pre-swizzled bf16 buffer: one contiguous 32B load per lane.
__device__ __forceinline__ bf16x16 load_b_swz(const unsigned short* __restrict__ swz,
                                              int kt, int nt, int ntiles, int lane) {
  const bf16x16* p = (const bf16x16*)(swz +
      (((long long)kt * ntiles + nt) * 32 + lane) * 16);
  return *p;
}

// A fragment from bf16 LDS tile, row-major [16 x stride].
__device__ __forceinline__ bf16x16 load_a_lds(const __bf16* h, int m, int half,
                                              int k0, int stride) {
  bf16x16 a;
#pragma unroll
  for (int e = 0; e < 16; ++e) a[e] = h[m * stride + k0 + a_kidx(e, half)];
  return a;
}

__device__ __forceinline__ float siluf(float v) { return v / (1.f + __expf(-v)); }
__device__ __forceinline__ float sigm(float v)  { return 1.f / (1.f + __expf(-v)); }

// ---------------------------------------------------------------------------
// Weight pre-swizzle: fp32 [K x ldn] row-major -> bf16 B-fragment order.
// out element idx = ((kt*ntiles + nt)*32 + lane)*16 + e ; k = kt*32+((e>>3)<<4)+half*8+(e&7),
// col = nt*16 + (lane&15). Zero-pads k >= K.
// ---------------------------------------------------------------------------
__global__ void k_swz_b(const float* __restrict__ W, unsigned short* __restrict__ out,
                        int K, int ldn, int ntiles, long long total) {
  long long idx = (long long)blockIdx.x * blockDim.x + threadIdx.x;
  if (idx >= total) return;
  int e    = (int)(idx & 15);
  int lane = (int)((idx >> 4) & 31);
  long long ft = idx >> 9;
  int nt = (int)(ft % ntiles);
  int kt = (int)(ft / ntiles);
  int half = lane >> 4;
  int col  = nt * 16 + (lane & 15);
  int k    = kt * 32 + ((e >> 3) << 4) + half * 8 + (e & 7);
  float v  = (k < K) ? W[(long long)k * ldn + col] : 0.f;
  __bf16 b = (__bf16)v;
  out[idx] = *(unsigned short*)&b;
}

// ---------------------------------------------------------------------------
// Node-feature tile loader into LDS (row-major [16 x cols]).
// ---------------------------------------------------------------------------
__device__ __forceinline__ void load_tile_lds(float* dst, const float* __restrict__ src,
                                              long long nb, long long N, int cols,
                                              int lane) {
  for (int i = lane; i < 16 * cols; i += 32) {
    int row = i / cols;
    int col = i - row * cols;
    long long n = nb + row;
    dst[i] = (n < N) ? src[n * cols + col] : 0.f;
  }
}

// ---------------------------------------------------------------------------
// Fused per-edge radial MLP: emb(16x10) -> silu(h 16x64) -> w (16xCK), WMMA.
// One wave handles 16 edges; results land in wlds (f32, row-major 16xCK).
// Basis recomputed from edge_len (no E x 10 buffer traffic).
// ---------------------------------------------------------------------------
template <int CK>
__device__ void radial16(const unsigned short* __restrict__ rW1s,
                         const float* __restrict__ rb1,
                         const unsigned short* __restrict__ rW2s,
                         const float* __restrict__ elen,
                         const float* __restrict__ rmax, long long ebase, long long E,
                         __bf16* hlds, float* wlds, int lane) {
  int m = lane & 15, half = lane >> 4, cidx = lane & 15;
  long long e = ebase + m;
  if (e >= E) e = E - 1;
  float d    = elen[e];
  float step = rmax[0] * (1.f / 9.f);
  float t    = d / step;

  bf16x16 a0;
#pragma unroll
  for (int i = 0; i < 16; ++i) {
    int k   = a_kidx(i, half);
    float v = 0.f;
    if (k < 10) {
      float df = t - (float)k;
      v = __expf(-df * df) * (1.f / 1.12f);
    }
    a0[i] = (__bf16)v;
  }

  floatx8 h[4] = {zero8(), zero8(), zero8(), zero8()};
#pragma unroll
  for (int nt = 0; nt < 4; ++nt) {
    bf16x16 b = load_b_swz(rW1s, 0, nt, 4, lane);
    h[nt] = wmma_bf16(a0, b, h[nt]);
  }
#pragma unroll
  for (int nt = 0; nt < 4; ++nt) {
    int col  = nt * 16 + cidx;
    float bb = rb1[col];
#pragma unroll
    for (int r = 0; r < 8; ++r)
      hlds[(half * 8 + r) * 64 + col] = (__bf16)siluf(h[nt][r] + bb);
  }
  __syncthreads();

  floatx8 w[CK / 16];
#pragma unroll
  for (int nt = 0; nt < CK / 16; ++nt) w[nt] = zero8();
#pragma unroll
  for (int kt = 0; kt < 2; ++kt) {
    bf16x16 a = load_a_lds(hlds, m, half, kt * 32, 64);
#pragma unroll
    for (int nt = 0; nt < CK / 16; ++nt) {
      bf16x16 b = load_b_swz(rW2s, kt, nt, CK / 16, lane);
      w[nt] = wmma_bf16(a, b, w[nt]);
    }
  }
#pragma unroll
  for (int nt = 0; nt < CK / 16; ++nt) {
    int col = nt * 16 + cidx;
#pragma unroll
    for (int r = 0; r < 8; ++r) wlds[(half * 8 + r) * CK + col] = w[nt][r];
  }
  __syncthreads();
}

// ---------------------------------------------------------------------------
// Embedding: xp = relu(relu(x)@W + b) @ lin1   (stage1 K=118->128, stage2 K=64)
// 8 waves / block, 16 nodes per wave.
// ---------------------------------------------------------------------------
__global__ void k_embed_x(const float* __restrict__ x,
                          const unsigned short* __restrict__ Wswz,
                          const float* __restrict__ bias,
                          const unsigned short* __restrict__ lin1swz,
                          float* __restrict__ xp, long long N) {
  __shared__ __bf16 xsL[8][16 * 64];
  int wave = threadIdx.x >> 5, lane = threadIdx.x & 31;
  int m = lane & 15, half = lane >> 4, cidx = lane & 15;
  long long nb = ((long long)blockIdx.x * 8 + wave) * 16;
  long long n0 = nb + m;
  if (n0 >= N) n0 = N - 1;

  floatx8 acc[4] = {zero8(), zero8(), zero8(), zero8()};
  for (int kt = 0; kt < 4; ++kt) {
    int k0 = kt * 32;
    bf16x16 a;
#pragma unroll
    for (int i = 0; i < 16; ++i) {
      int k   = k0 + a_kidx(i, half);
      float v = (k < 118) ? fmaxf(x[n0 * 118 + k], 0.f) : 0.f;
      a[i] = (__bf16)v;
    }
#pragma unroll
    for (int nt = 0; nt < 4; ++nt) {
      bf16x16 b = load_b_swz(Wswz, kt, nt, 4, lane);
      acc[nt] = wmma_bf16(a, b, acc[nt]);
    }
  }
  __bf16* xsw = xsL[wave];
#pragma unroll
  for (int nt = 0; nt < 4; ++nt) {
    int col  = nt * 16 + cidx;
    float bb = bias[col];
#pragma unroll
    for (int r = 0; r < 8; ++r)
      xsw[(half * 8 + r) * 64 + col] = (__bf16)fmaxf(acc[nt][r] + bb, 0.f);
  }
  __syncthreads();

  floatx8 p[2] = {zero8(), zero8()};
#pragma unroll
  for (int kt = 0; kt < 2; ++kt) {
    bf16x16 a = load_a_lds(xsw, m, half, kt * 32, 64);
#pragma unroll
    for (int nt = 0; nt < 2; ++nt) {
      bf16x16 b = load_b_swz(lin1swz, kt, nt, 2, lane);
      p[nt] = wmma_bf16(a, b, p[nt]);
    }
  }
#pragma unroll
  for (int nt = 0; nt < 2; ++nt) {
    int col = nt * 16 + cidx;
#pragma unroll
    for (int r = 0; r < 8; ++r) {
      long long n = nb + half * 8 + r;
      if (n < N) xp[n * 32 + col] = p[nt][r];
    }
  }
}

__global__ void k_embed_z(const float* __restrict__ z,
                          const unsigned short* __restrict__ Wswz,
                          const float* __restrict__ bias, float* __restrict__ zz,
                          long long N) {
  int wave = threadIdx.x >> 5, lane = threadIdx.x & 31;
  int m = lane & 15, half = lane >> 4, cidx = lane & 15;
  long long nb = ((long long)blockIdx.x * 8 + wave) * 16;
  long long n0 = nb + m;
  if (n0 >= N) n0 = N - 1;

  floatx8 acc[4] = {zero8(), zero8(), zero8(), zero8()};
  for (int kt = 0; kt < 4; ++kt) {
    int k0 = kt * 32;
    bf16x16 a;
#pragma unroll
    for (int i = 0; i < 16; ++i) {
      int k   = k0 + a_kidx(i, half);
      float v = (k < 118) ? fmaxf(z[n0 * 118 + k], 0.f) : 0.f;
      a[i] = (__bf16)v;
    }
#pragma unroll
    for (int nt = 0; nt < 4; ++nt) {
      bf16x16 b = load_b_swz(Wswz, kt, nt, 4, lane);
      acc[nt] = wmma_bf16(a, b, acc[nt]);
    }
  }
#pragma unroll
  for (int nt = 0; nt < 4; ++nt) {
    int col  = nt * 16 + cidx;
    float bb = bias[col];
#pragma unroll
    for (int r = 0; r < 8; ++r) {
      long long n = nb + half * 8 + r;
      if (n < N) zz[n * 64 + col] = fmaxf(acc[nt][r] + bb, 0.f);
    }
  }
}

// ---------------------------------------------------------------------------
// Edge kernels: 128 threads = 4 waves = 64 edges; WMMA radial + atomic scatter.
// ---------------------------------------------------------------------------
__global__ void k_edge_l0(const int* __restrict__ eidx, const float* __restrict__ evec,
                          const float* __restrict__ elen, const float* __restrict__ rmax,
                          const unsigned short* __restrict__ rW1s,
                          const float* __restrict__ rb1,
                          const unsigned short* __restrict__ rW2s,
                          const float* __restrict__ xp,
                          float* __restrict__ a_s, float* __restrict__ a_v, long long E) {
  __shared__ __bf16 hlds[4][16 * 64];
  __shared__ float  wlds[4][16 * 64];
  int wave = threadIdx.x >> 5, lane = threadIdx.x & 31;
  long long bb = (long long)blockIdx.x * 64;
  radial16<64>(rW1s, rb1, rW2s, elen, rmax, bb + wave * 16, E, hlds[wave], wlds[wave],
               lane);

  int t = threadIdx.x, le = t >> 1;
  long long e = bb + le;
  if (e < E) {
    int ch = (t & 1) * 16;
    long long src = eidx[e], dst = eidx[E + e];
    float inv = SQ3f / elen[e];
    float sh0 = evec[e * 3 + 0] * inv, sh1 = evec[e * 3 + 1] * inv,
          sh2 = evec[e * 3 + 2] * inv;
    const float* w = wlds[le >> 4] + (le & 15) * 64;
    for (int c = ch; c < ch + 16; ++c) {
      float g = xp[src * 32 + c];
      atomicAdd(&a_s[dst * 32 + c], w[c * 2 + 0] * g);
      float wv = w[c * 2 + 1] * g;
      atomicAdd(&a_v[dst * 96 + c * 3 + 0], wv * sh0);
      atomicAdd(&a_v[dst * 96 + c * 3 + 1], wv * sh1);
      atomicAdd(&a_v[dst * 96 + c * 3 + 2], wv * sh2);
    }
  }
}

__global__ void k_edge_l1(const int* __restrict__ eidx, const float* __restrict__ evec,
                          const float* __restrict__ elen, const float* __restrict__ rmax,
                          const unsigned short* __restrict__ rW1s,
                          const float* __restrict__ rb1,
                          const unsigned short* __restrict__ rW2s,
                          const float* __restrict__ xps,
                          const float* __restrict__ xpv, float* __restrict__ a_s,
                          float* __restrict__ a_v, long long E) {
  __shared__ __bf16 hlds[4][16 * 64];
  __shared__ float  wlds[4][16 * 160];
  int wave = threadIdx.x >> 5, lane = threadIdx.x & 31;
  long long bb = (long long)blockIdx.x * 64;
  radial16<160>(rW1s, rb1, rW2s, elen, rmax, bb + wave * 16, E, hlds[wave], wlds[wave],
                lane);

  int t = threadIdx.x, le = t >> 1;
  long long e = bb + le;
  if (e < E) {
    int ch = (t & 1) * 16;
    long long src = eidx[e], dst = eidx[E + e];
    float inv = SQ3f / elen[e];
    float sh0 = evec[e * 3 + 0] * inv, sh1 = evec[e * 3 + 1] * inv,
          sh2 = evec[e * 3 + 2] * inv;
    const float* w = wlds[le >> 4] + (le & 15) * 160;
    for (int c = ch; c < ch + 16; ++c) {
      const float* wc = w + c * 5;
      float gs = xps[src * 32 + c];
      float g0 = xpv[src * 96 + c * 3 + 0];
      float g1 = xpv[src * 96 + c * 3 + 1];
      float g2 = xpv[src * 96 + c * 3 + 2];
      float dot = (g0 * sh0 + g1 * sh1 + g2 * sh2) * INV_SQ3f;
      float cr0 = (g1 * sh2 - g2 * sh1) * INV_SQ2f;
      float cr1 = (g2 * sh0 - g0 * sh2) * INV_SQ2f;
      float cr2 = (g0 * sh1 - g1 * sh0) * INV_SQ2f;
      atomicAdd(&a_s[dst * 32 + c], wc[0] * gs + wc[3] * dot);
      float w1gs = wc[1] * gs;
      atomicAdd(&a_v[dst * 96 + c * 3 + 0], w1gs * sh0 + wc[2] * g0 + wc[4] * cr0);
      atomicAdd(&a_v[dst * 96 + c * 3 + 1], w1gs * sh1 + wc[2] * g1 + wc[4] * cr1);
      atomicAdd(&a_v[dst * 96 + c * 3 + 2], w1gs * sh2 + wc[2] * g2 + wc[4] * cr2);
    }
  }
}

__global__ void k_edge_lf(const int* __restrict__ eidx, const float* __restrict__ evec,
                          const float* __restrict__ elen, const float* __restrict__ rmax,
                          const unsigned short* __restrict__ rW1s,
                          const float* __restrict__ rb1,
                          const unsigned short* __restrict__ rW2s,
                          const float* __restrict__ xps,
                          const float* __restrict__ xpv, float* __restrict__ a_s,
                          long long E) {
  __shared__ __bf16 hlds[4][16 * 64];
  __shared__ float  wlds[4][16 * 64];
  int wave = threadIdx.x >> 5, lane = threadIdx.x & 31;
  long long bb = (long long)blockIdx.x * 64;
  radial16<64>(rW1s, rb1, rW2s, elen, rmax, bb + wave * 16, E, hlds[wave], wlds[wave],
               lane);

  int t = threadIdx.x, le = t >> 1;
  long long e = bb + le;
  if (e < E) {
    int ch = (t & 1) * 16;
    long long src = eidx[e], dst = eidx[E + e];
    float inv = SQ3f / elen[e];
    float sh0 = evec[e * 3 + 0] * inv, sh1 = evec[e * 3 + 1] * inv,
          sh2 = evec[e * 3 + 2] * inv;
    const float* w = wlds[le >> 4] + (le & 15) * 64;
    for (int c = ch; c < ch + 16; ++c) {
      float gs  = xps[src * 32 + c];
      float dot = (xpv[src * 96 + c * 3 + 0] * sh0 + xpv[src * 96 + c * 3 + 1] * sh1 +
                   xpv[src * 96 + c * 3 + 2] * sh2) * INV_SQ3f;
      atomicAdd(&a_s[dst * 32 + c], w[c * 2 + 0] * gs + w[c * 2 + 1] * dot);
    }
  }
}

// ---------------------------------------------------------------------------
// Node kernels: 64 threads = 2 waves = 32 nodes; WMMA einsum K=2048 + scalar mix.
// ---------------------------------------------------------------------------
__global__ void k_node_l0(const float* __restrict__ a_s, const float* __restrict__ a_v,
                          const float* __restrict__ deg, const float* __restrict__ xp,
                          const float* __restrict__ zz, const float* __restrict__ W2s,
                          const float* __restrict__ W2v,
                          const unsigned short* __restrict__ Wscs,
                          const float* __restrict__ nxt_lin_s,
                          const float* __restrict__ nxt_lin_v,
                          float* __restrict__ xps_out, float* __restrict__ xpv_out,
                          long long N) {
  __shared__ float xpL[2][16 * 32];
  __shared__ float zzL[2][16 * 64];
  __shared__ float osL[2][16 * 64];
  __shared__ float ovL[2][16 * 96];
  int wave = threadIdx.x >> 5, lane = threadIdx.x & 31;
  int m = lane & 15, half = lane >> 4, cidx = lane & 15;
  long long nb = ((long long)blockIdx.x * 2 + wave) * 16;
  float *xpw = xpL[wave], *zzw = zzL[wave], *osw = osL[wave], *ovw = ovL[wave];

  load_tile_lds(xpw, xp, nb, N, 32, lane);
  load_tile_lds(zzw, zz, nb, N, 64, lane);
  __syncthreads();

  floatx8 acc[4] = {zero8(), zero8(), zero8(), zero8()};
  for (int kt = 0; kt < 64; ++kt) {
    if (kt < 63) __builtin_prefetch(&Wscs[((long long)(kt + 1) * 4) * 512]);
    bf16x16 a;
#pragma unroll
    for (int i = 0; i < 16; ++i) {
      int k = kt * 32 + a_kidx(i, half);
      a[i] = (__bf16)(xpw[m * 32 + (k >> 6)] * zzw[m * 64 + (k & 63)]);
    }
#pragma unroll
    for (int nt = 0; nt < 4; ++nt) {
      bf16x16 b = load_b_swz(Wscs, kt, nt, 4, lane);
      acc[nt] = wmma_bf16(a, b, acc[nt]);
    }
  }
#pragma unroll
  for (int nt = 0; nt < 4; ++nt) {
    int col = nt * 16 + cidx;
#pragma unroll
    for (int r = 0; r < 8; ++r) osw[(half * 8 + r) * 64 + col] = acc[nt][r];
  }
  __syncthreads();

  int mn = lane >> 1, h = lane & 1;
  long long n = nb + mn;
  if (n < N) {
    float dinv = 1.f / deg[n];
    for (int o = h * 32; o < h * 32 + 32; ++o) {
      float s = osw[mn * 64 + o];
      for (int c = 0; c < 32; ++c) s += a_s[n * 32 + c] * dinv * W2s[c * 64 + o];
      osw[mn * 64 + o] = s;
    }
    for (int o = h * 16; o < h * 16 + 16; ++o)
      for (int d = 0; d < 3; ++d) {
        float s = 0.f;
        for (int c = 0; c < 32; ++c) s += a_v[n * 96 + c * 3 + d] * dinv * W2v[c * 32 + o];
        ovw[mn * 96 + o * 3 + d] = s;
      }
  }
  __syncthreads();
  if (n < N) {
    for (int o = h * 16; o < h * 16 + 16; ++o) {
      float s = 0.f, s0 = 0.f, s1 = 0.f, s2 = 0.f;
      for (int c = 0; c < 32; ++c) {
        float fs   = siluf(osw[mn * 64 + c]);
        float gate = sigm(osw[mn * 64 + 32 + c]);
        s += fs * nxt_lin_s[c * 32 + o];
        float wv = gate * nxt_lin_v[c * 32 + o];
        s0 += wv * ovw[mn * 96 + c * 3 + 0];
        s1 += wv * ovw[mn * 96 + c * 3 + 1];
        s2 += wv * ovw[mn * 96 + c * 3 + 2];
      }
      xps_out[n * 32 + o] = s;
      xpv_out[n * 96 + o * 3 + 0] = s0;
      xpv_out[n * 96 + o * 3 + 1] = s1;
      xpv_out[n * 96 + o * 3 + 2] = s2;
    }
  }
}

__global__ void k_node_l1(const float* __restrict__ a_s, const float* __restrict__ a_v,
                          const float* __restrict__ deg, const float* __restrict__ xps,
                          const float* __restrict__ zz, const float* __restrict__ xpv,
                          const float* __restrict__ W2s, const float* __restrict__ W2v,
                          const unsigned short* __restrict__ Wscs,
                          const unsigned short* __restrict__ Wscv,
                          const float* __restrict__ nxt_lin_s,
                          const float* __restrict__ nxt_lin_v,
                          float* __restrict__ xps_out, float* __restrict__ xpv_out,
                          long long N) {
  __shared__ float xsL[2][16 * 32];
  __shared__ float zzL[2][16 * 64];
  __shared__ float xvL[2][16 * 96];  // holds xpv, then reused for o_v
  __shared__ float osL[2][16 * 64];
  int wave = threadIdx.x >> 5, lane = threadIdx.x & 31;
  int m = lane & 15, half = lane >> 4, cidx = lane & 15;
  long long nb = ((long long)blockIdx.x * 2 + wave) * 16;
  float *xsw = xsL[wave], *zzw = zzL[wave], *xvw = xvL[wave], *osw = osL[wave];

  load_tile_lds(xsw, xps, nb, N, 32, lane);
  load_tile_lds(zzw, zz, nb, N, 64, lane);
  load_tile_lds(xvw, xpv, nb, N, 96, lane);
  __syncthreads();

  floatx8 accS[4] = {zero8(), zero8(), zero8(), zero8()};
  floatx8 accV[3][2];
#pragma unroll
  for (int d = 0; d < 3; ++d)
    for (int nt = 0; nt < 2; ++nt) accV[d][nt] = zero8();

  for (int kt = 0; kt < 64; ++kt) {
    if (kt < 63) {
      __builtin_prefetch(&Wscs[((long long)(kt + 1) * 4) * 512]);
      __builtin_prefetch(&Wscv[((long long)(kt + 1) * 2) * 512]);
    }
    bf16x16 aS, aV0, aV1, aV2;
#pragma unroll
    for (int i = 0; i < 16; ++i) {
      int k = kt * 32 + a_kidx(i, half);
      int c = k >> 6, zi = k & 63;
      float zv = zzw[m * 64 + zi];
      aS[i]  = (__bf16)(xsw[m * 32 + c] * zv);
      aV0[i] = (__bf16)(xvw[m * 96 + c * 3 + 0] * zv);
      aV1[i] = (__bf16)(xvw[m * 96 + c * 3 + 1] * zv);
      aV2[i] = (__bf16)(xvw[m * 96 + c * 3 + 2] * zv);
    }
#pragma unroll
    for (int nt = 0; nt < 4; ++nt) {
      bf16x16 b = load_b_swz(Wscs, kt, nt, 4, lane);
      accS[nt] = wmma_bf16(aS, b, accS[nt]);
    }
#pragma unroll
    for (int nt = 0; nt < 2; ++nt) {
      bf16x16 b = load_b_swz(Wscv, kt, nt, 2, lane);
      accV[0][nt] = wmma_bf16(aV0, b, accV[0][nt]);
      accV[1][nt] = wmma_bf16(aV1, b, accV[1][nt]);
      accV[2][nt] = wmma_bf16(aV2, b, accV[2][nt]);
    }
  }
#pragma unroll
  for (int nt = 0; nt < 4; ++nt) {
    int col = nt * 16 + cidx;
#pragma unroll
    for (int r = 0; r < 8; ++r) osw[(half * 8 + r) * 64 + col] = accS[nt][r];
  }
  // xvw reads are done (lockstep within wave) -> reuse as o_v tile
#pragma unroll
  for (int nt = 0; nt < 2; ++nt) {
    int col = nt * 16 + cidx;
#pragma unroll
    for (int r = 0; r < 8; ++r) {
      int row = half * 8 + r;
      xvw[row * 96 + col * 3 + 0] = accV[0][nt][r];
      xvw[row * 96 + col * 3 + 1] = accV[1][nt][r];
      xvw[row * 96 + col * 3 + 2] = accV[2][nt][r];
    }
  }
  __syncthreads();

  int mn = lane >> 1, h = lane & 1;
  long long n = nb + mn;
  if (n < N) {
    float dinv = 1.f / deg[n];
    for (int o = h * 32; o < h * 32 + 32; ++o) {
      float s = osw[mn * 64 + o];
      for (int c = 0; c < 32; ++c) s += a_s[n * 32 + c] * dinv * W2s[c * 64 + o];
      osw[mn * 64 + o] = s;
    }
    for (int o = h * 16; o < h * 16 + 16; ++o)
      for (int d = 0; d < 3; ++d) {
        float s = xvw[mn * 96 + o * 3 + d];
        for (int c = 0; c < 32; ++c) s += a_v[n * 96 + c * 3 + d] * dinv * W2v[c * 32 + o];
        xvw[mn * 96 + o * 3 + d] = s;
      }
  }
  __syncthreads();
  if (n < N) {
    for (int o = h * 16; o < h * 16 + 16; ++o) {
      float s = 0.f, s0 = 0.f, s1 = 0.f, s2 = 0.f;
      for (int c = 0; c < 32; ++c) {
        float fs   = siluf(osw[mn * 64 + c]);
        float gate = sigm(osw[mn * 64 + 32 + c]);
        s += fs * nxt_lin_s[c * 32 + o];
        float wv = gate * nxt_lin_v[c * 32 + o];
        s0 += wv * xvw[mn * 96 + c * 3 + 0];
        s1 += wv * xvw[mn * 96 + c * 3 + 1];
        s2 += wv * xvw[mn * 96 + c * 3 + 2];
      }
      xps_out[n * 32 + o] = s;
      xpv_out[n * 96 + o * 3 + 0] = s0;
      xpv_out[n * 96 + o * 3 + 1] = s1;
      xpv_out[n * 96 + o * 3 + 2] = s2;
    }
  }
}

__global__ void k_node_lf(const float* __restrict__ a_s, const float* __restrict__ deg,
                          const float* __restrict__ xps, const float* __restrict__ zz,
                          const float* __restrict__ W2,
                          const unsigned short* __restrict__ Wscs,
                          const int* __restrict__ batch, float* __restrict__ pooled,
                          float* __restrict__ cnt, long long N) {
  __shared__ float xsL[2][16 * 32];
  __shared__ float zzL[2][16 * 64];
  __shared__ float osL[2][16 * 64];
  int wave = threadIdx.x >> 5, lane = threadIdx.x & 31;
  int m = lane & 15, half = lane >> 4, cidx = lane & 15;
  long long nb = ((long long)blockIdx.x * 2 + wave) * 16;
  float *xsw = xsL[wave], *zzw = zzL[wave], *osw = osL[wave];

  load_tile_lds(xsw, xps, nb, N, 32, lane);
  load_tile_lds(zzw, zz, nb, N, 64, lane);
  __syncthreads();

  floatx8 acc[4] = {zero8(), zero8(), zero8(), zero8()};
  for (int kt = 0; kt < 64; ++kt) {
    if (kt < 63) __builtin_prefetch(&Wscs[((long long)(kt + 1) * 4) * 512]);
    bf16x16 a;
#pragma unroll
    for (int i = 0; i < 16; ++i) {
      int k = kt * 32 + a_kidx(i, half);
      a[i] = (__bf16)(xsw[m * 32 + (k >> 6)] * zzw[m * 64 + (k & 63)]);
    }
#pragma unroll
    for (int nt = 0; nt < 4; ++nt) {
      bf16x16 b = load_b_swz(Wscs, kt, nt, 4, lane);
      acc[nt] = wmma_bf16(a, b, acc[nt]);
    }
  }
#pragma unroll
  for (int nt = 0; nt < 4; ++nt) {
    int col = nt * 16 + cidx;
#pragma unroll
    for (int r = 0; r < 8; ++r) osw[(half * 8 + r) * 64 + col] = acc[nt][r];
  }
  __syncthreads();

  int mn = lane >> 1, h = lane & 1;
  long long n = nb + mn;
  if (n < N) {
    float dinv = 1.f / deg[n];
    int g = batch[n];
    for (int o = h * 32; o < h * 32 + 32; ++o) {
      float s = osw[mn * 64 + o];
      for (int c = 0; c < 32; ++c) s += a_s[n * 32 + c] * dinv * W2[c * 64 + o];
      atomicAdd(&pooled[(long long)g * 64 + o], s);
    }
    if (h == 0) atomicAdd(&cnt[g], 1.f);
  }
}

__global__ void k_final(const float* __restrict__ pooled, const float* __restrict__ cnt,
                        const float* __restrict__ clsW, const float* __restrict__ clsb,
                        float* __restrict__ out, int G) {
  int g = blockIdx.x * blockDim.x + threadIdx.x;
  if (g < G) {
    float c = fmaxf(cnt[g], 1.f);
    float s = clsb[0];
    for (int o = 0; o < 64; ++o) s += (pooled[(long long)g * 64 + o] / c) * clsW[o];
    out[g] = sigm(s);
  }
}

// ---------------------------------------------------------------------------
// Host orchestration (graph-capture safe: only async memset + kernel launches)
// ---------------------------------------------------------------------------
extern "C" void kernel_launch(void* const* d_in, const int* in_sizes, int n_in,
                              void* d_out, int out_size, void* d_ws, size_t ws_size,
                              hipStream_t stream) {
  const float* x     = (const float*)d_in[0];
  const float* z     = (const float*)d_in[1];
  const int*   eidx  = (const int*)d_in[2];
  const float* evec  = (const float*)d_in[3];
  const float* elen  = (const float*)d_in[4];
  const float* deg   = (const float*)d_in[5];
  const int*   batch = (const int*)d_in[6];
  const float* rmax  = (const float*)d_in[7];
  const float* emx_W = (const float*)d_in[8];
  const float* emx_b = (const float*)d_in[9];
  const float* emz_W = (const float*)d_in[10];
  const float* emz_b = (const float*)d_in[11];
  const float* l0_lin1_s = (const float*)d_in[12];
  const float* l0_rW1 = (const float*)d_in[13];
  const float* l0_rb1 = (const float*)d_in[14];
  const float* l0_rW2 = (const float*)d_in[15];
  const float* l0_W2s = (const float*)d_in[16];
  const float* l0_W2v = (const float*)d_in[17];
  const float* l0_Wsc_s = (const float*)d_in[18];
  const float* l1_lin1_s = (const float*)d_in[19];
  const float* l1_lin1_v = (const float*)d_in[20];
  const float* l1_rW1 = (const float*)d_in[21];
  const float* l1_rb1 = (const float*)d_in[22];
  const float* l1_rW2 = (const float*)d_in[23];
  const float* l1_W2s = (const float*)d_in[24];
  const float* l1_W2v = (const float*)d_in[25];
  const float* l1_Wsc_s = (const float*)d_in[26];
  const float* l1_Wsc_v = (const float*)d_in[27];
  const float* lf_lin1_s = (const float*)d_in[28];
  const float* lf_lin1_v = (const float*)d_in[29];
  const float* lf_rW1 = (const float*)d_in[30];
  const float* lf_rb1 = (const float*)d_in[31];
  const float* lf_rW2 = (const float*)d_in[32];
  const float* lf_W2  = (const float*)d_in[33];
  const float* lf_Wsc = (const float*)d_in[34];
  const float* cls_W  = (const float*)d_in[35];
  const float* cls_b  = (const float*)d_in[36];

  long long N = in_sizes[5];  // node_deg has N elements
  long long E = in_sizes[4];  // edge_len has E elements
  int G = out_size;
  (void)n_in; (void)ws_size;

  char* ws = (char*)d_ws;
  size_t off = 0;
  auto carve = [&](size_t bytes) -> void* {
    void* p = ws + off;
    off = (off + bytes + 255) & ~(size_t)255;
    return p;
  };
  float* xpA = (float*)carve((size_t)N * 32 * 4);
  float* xvA = (float*)carve((size_t)N * 96 * 4);
  float* xpB = (float*)carve((size_t)N * 32 * 4);
  float* xvB = (float*)carve((size_t)N * 96 * 4);
  float* zz  = (float*)carve((size_t)N * 64 * 4);
  float* a_s = (float*)carve((size_t)N * 32 * 4);
  float* a_v = (float*)carve((size_t)N * 96 * 4);
  float* pooled = (float*)carve((size_t)G * 64 * 4);
  float* cnt    = (float*)carve((size_t)G * 4);

  // Pre-swizzle all WMMA B-operand weights into bf16 fragment order.
  auto swz = [&](const float* W, int K, int ldn) -> unsigned short* {
    int ktiles = (K + 31) / 32, ntiles = ldn / 16;
    long long total = (long long)ktiles * ntiles * 512;
    unsigned short* o = (unsigned short*)carve((size_t)total * 2);
    int blocks = (int)((total + 255) / 256);
    k_swz_b<<<blocks, 256, 0, stream>>>(W, o, K, ldn, ntiles, total);
    return o;
  };
  unsigned short* s_emx_W    = swz(emx_W, 118, 64);
  unsigned short* s_emz_W    = swz(emz_W, 118, 64);
  unsigned short* s_l0_lin1  = swz(l0_lin1_s, 64, 32);
  unsigned short* s_l0_rW1   = swz(l0_rW1, 10, 64);
  unsigned short* s_l0_rW2   = swz(l0_rW2, 64, 64);
  unsigned short* s_l1_rW1   = swz(l1_rW1, 10, 64);
  unsigned short* s_l1_rW2   = swz(l1_rW2, 64, 160);
  unsigned short* s_lf_rW1   = swz(lf_rW1, 10, 64);
  unsigned short* s_lf_rW2   = swz(lf_rW2, 64, 64);
  unsigned short* s_l0_Wsc_s = swz(l0_Wsc_s, 2048, 64);
  unsigned short* s_l1_Wsc_s = swz(l1_Wsc_s, 2048, 64);
  unsigned short* s_l1_Wsc_v = swz(l1_Wsc_v, 2048, 32);
  unsigned short* s_lf_Wsc   = swz(lf_Wsc, 2048, 64);

  int embBlocks  = (int)((N + 127) / 128);
  int nodeBlocks = (int)((N + 31) / 32);
  int edgeBlocks = (int)((E + 63) / 64);

  k_embed_x<<<embBlocks, 256, 0, stream>>>(x, s_emx_W, emx_b, s_l0_lin1, xpA, N);
  k_embed_z<<<embBlocks, 256, 0, stream>>>(z, s_emz_W, emz_b, zz, N);

  // ---- layer 0 ----
  hipMemsetAsync(a_s, 0, (size_t)N * 32 * 4, stream);
  hipMemsetAsync(a_v, 0, (size_t)N * 96 * 4, stream);
  k_edge_l0<<<edgeBlocks, 128, 0, stream>>>(eidx, evec, elen, rmax, s_l0_rW1, l0_rb1,
                                            s_l0_rW2, xpA, a_s, a_v, E);
  k_node_l0<<<nodeBlocks, 64, 0, stream>>>(a_s, a_v, deg, xpA, zz, l0_W2s, l0_W2v,
                                           s_l0_Wsc_s, l1_lin1_s, l1_lin1_v, xpB, xvB,
                                           N);

  // ---- layer 1 ----
  hipMemsetAsync(a_s, 0, (size_t)N * 32 * 4, stream);
  hipMemsetAsync(a_v, 0, (size_t)N * 96 * 4, stream);
  k_edge_l1<<<edgeBlocks, 128, 0, stream>>>(eidx, evec, elen, rmax, s_l1_rW1, l1_rb1,
                                            s_l1_rW2, xpB, xvB, a_s, a_v, E);
  k_node_l1<<<nodeBlocks, 64, 0, stream>>>(a_s, a_v, deg, xpB, zz, xvB, l1_W2s, l1_W2v,
                                           s_l1_Wsc_s, s_l1_Wsc_v, lf_lin1_s, lf_lin1_v,
                                           xpA, xvA, N);

  // ---- final layer + pooling ----
  hipMemsetAsync(a_s, 0, (size_t)N * 32 * 4, stream);
  hipMemsetAsync(pooled, 0, (size_t)G * 64 * 4, stream);
  hipMemsetAsync(cnt, 0, (size_t)G * 4, stream);
  k_edge_lf<<<edgeBlocks, 128, 0, stream>>>(eidx, evec, elen, rmax, s_lf_rW1, lf_rb1,
                                            s_lf_rW2, xpA, xvA, a_s, E);
  k_node_lf<<<nodeBlocks, 64, 0, stream>>>(a_s, deg, xpA, zz, lf_W2, s_lf_Wsc, batch,
                                           pooled, cnt, N);
  k_final<<<(G + 255) / 256, 256, 0, stream>>>(pooled, cnt, cls_W, cls_b,
                                               (float*)d_out, G);
}
